// MultiHeadAttention_17798344474881
// MI455X (gfx1250) — matrix operand order repaired and checked by reading, not compiled, not run on hardware
//
#include <hip/hip_runtime.h>
#include <hip/hip_bf16.h>
#include <stddef.h>

// ---------------- problem constants ----------------
#define DMODEL 768
#define NHEAD  12
#define SEQ    1024
#define BATCH  6
#define DHEAD  64
#define NT     (BATCH*SEQ)             // 6144 total rows
#define WELEM  (DMODEL*DMODEL)         // 589824 elements per weight
#define QKVE   (BATCH*NHEAD*SEQ*DHEAD) // 4718592 elements per q/k/v tensor

// ---------------- WMMA types ----------------
typedef __attribute__((ext_vector_type(16))) __bf16 v16bf;
typedef __attribute__((ext_vector_type(8)))  __bf16 v8bf;
typedef __attribute__((ext_vector_type(8)))  float  v8f;

union BFrag {
    v16bf  v;
    v8bf   h[2];
    __bf16 e[16];
};

static __device__ __forceinline__ v8f wmma_bf16(v16bf a, v16bf b, v8f c) {
    return __builtin_amdgcn_wmma_f32_16x16x32_bf16(
        /*neg_a=*/false, a, /*neg_b=*/false, b,
        /*c_mod=*/(short)0, c, /*reuse_a=*/false, /*reuse_b=*/false);
}

// A fragment (16x32 bf16): per lane 8 contiguous + 8 contiguous halves.
// p must already point at row_base + K0 + (laneHi ? 8 : 0).
static __device__ __forceinline__ v16bf load_a_bf16(const __bf16* p) {
    BFrag f;
    f.h[0] = *(const v8bf*)(p);
    f.h[1] = *(const v8bf*)(p + 16);
    return f.v;
}

// A fragment built from f32 source (native cast -> v_cvt bf16).
static __device__ __forceinline__ v16bf load_a_f32(const float* p) {
    BFrag f;
#pragma unroll
    for (int j = 0; j < 8; ++j) {
        f.e[j]     = (__bf16)p[j];
        f.e[j + 8] = (__bf16)p[16 + j];
    }
    return f.v;
}

// B fragment (32x16 bf16): per lane 8 contiguous + 8 contiguous halves (fixed column).
// p must already point at col_base + K0 + (laneHi ? 8 : 0).
static __device__ __forceinline__ v16bf load_b_bf16(const __bf16* p) {
    BFrag f;
    f.h[0] = *(const v8bf*)(p);
    f.h[1] = *(const v8bf*)(p + 16);
    return f.v;
}

static __device__ __forceinline__ v8f vzero8() {
    v8f z = {0.f, 0.f, 0.f, 0.f, 0.f, 0.f, 0.f, 0.f};
    return z;
}

// ============================================================
// Kernel 0: transpose + convert the four 768x768 f32 weights to
// bf16 W^T[n][k] so WMMA B-fragments are contiguous along K.
// ============================================================
__global__ __launch_bounds__(256) void prep_weights_kernel(
    const float* __restrict__ wq, const float* __restrict__ wk,
    const float* __restrict__ wv, const float* __restrict__ wo,
    __bf16* __restrict__ out)
{
    int idx = blockIdx.x * 256 + threadIdx.x;        // 4*WELEM total
    int which = idx / WELEM;
    int rem   = idx - which * WELEM;
    int n = rem / DMODEL;
    int k = rem - n * DMODEL;
    const float* W = (which == 0) ? wq : (which == 1) ? wk : (which == 2) ? wv : wo;
    out[(size_t)which * WELEM + (size_t)n * DMODEL + k] =
        (__bf16)W[(size_t)k * DMODEL + n];
}

// ============================================================
// Kernel 1: QKV projection.  grid = (NT/64, 768/64, 3)
// slot 0: q -> bf16 [B,H,S,DK] ; slot 1: k -> bf16 [B,H,S,DK]
// slot 2: v -> bf16 [B,H,DK,S]  (transposed for context B-frags)
// ============================================================
__global__ __launch_bounds__(256) void qkv_proj_kernel(
    const float* __restrict__ Q, const float* __restrict__ K,
    const float* __restrict__ V,
    const __bf16* __restrict__ wt,           // 3 transposed bf16 weights
    const float* __restrict__ bq, const float* __restrict__ bk,
    const float* __restrict__ bv,
    __bf16* __restrict__ qkv_out)            // q | k | vT, QKVE elems each
{
    const int slot = blockIdx.z;
    const float* X = (slot == 0) ? Q : (slot == 1) ? K : V;
    const __bf16* W = wt + (size_t)slot * WELEM;
    const float* bias = (slot == 0) ? bq : (slot == 1) ? bk : bv;
    __bf16* out = qkv_out + (size_t)slot * QKVE;

    const int tid  = threadIdx.x;
    const int lane = tid & 31;
    const int wave = tid >> 5;
    const int lrow = lane & 15;
    const int hi8  = (lane & 16) ? 8 : 0;
    const int hi16 = (lane & 16) ? 16 : 0;

    // 16 (4x4) 16x16 tiles per block, 2 per wave sharing the A row-tile.
    const int t0  = wave * 2;
    const int mi  = t0 >> 2;
    const int ni0 = t0 & 3;

    const int arow = blockIdx.x * 64 + mi * 16 + lrow;
    const int n0   = blockIdx.y * 64 + ni0 * 16 + lrow;
    const int n1   = n0 + 16;

    v8f c0 = vzero8(), c1 = vzero8();
    const float* ap = X + (size_t)arow * DMODEL + hi8;
    const __bf16* bp0 = W + (size_t)n0 * DMODEL + hi8;
    const __bf16* bp1 = W + (size_t)n1 * DMODEL + hi8;

#pragma unroll 4
    for (int kc = 0; kc < 24; ++kc) {
        const int k0 = kc * 32;
        v16bf a  = load_a_f32(ap + k0);
        v16bf b0 = load_b_bf16(bp0 + k0);
        v16bf b1 = load_b_bf16(bp1 + k0);
        c0 = wmma_bf16(a, b0, c0);
        c1 = wmma_bf16(a, b1, c1);
    }

    const int Mbase = blockIdx.x * 64 + mi * 16 + ((lane & 16) ? 8 : 0);
#pragma unroll
    for (int t = 0; t < 2; ++t) {
        v8f cc = t ? c1 : c0;
        int col = t ? n1 : n0;
        float bvl = bias[col];
        int h = col >> 6, d = col & 63;
#pragma unroll
        for (int r = 0; r < 8; ++r) {
            int trow = Mbase + r;
            int b = trow >> 10, s = trow & 1023;
            __bf16 val = (__bf16)(cc[r] + bvl);
            if (slot < 2)
                out[(((size_t)(b * NHEAD + h) * SEQ + s) * DHEAD) + d] = val;
            else
                out[(((size_t)(b * NHEAD + h) * DHEAD + d) * SEQ) + s] = val;
        }
    }
}

// ============================================================
// Kernel 2: attention.  grid = (SEQ/16, B*H), 256 threads,
// dynamic LDS = 64KB f32 scores + 32KB bf16 probs = 96KB.
// ============================================================
__global__ __launch_bounds__(256) void attn_kernel(
    const __bf16* __restrict__ qbf,
    const __bf16* __restrict__ kbf,
    const __bf16* __restrict__ vtbf,
    const unsigned char* __restrict__ mask,
    float* __restrict__ attn_out,
    __bf16* __restrict__ ctx)
{
    extern __shared__ char smem[];
    float*  sc = (float*)smem;                        // [16][1024] scores
    __bf16* pb = (__bf16*)(smem + 16 * SEQ * 4);      // [16][1024] probs bf16
    __shared__ float red[256];
    __shared__ float psum[8 * 16 * 16];               // partial context tiles

    const int bh = blockIdx.y;
    const int b  = bh / NHEAD;
    const int h  = bh - b * NHEAD;
    const int qbase = blockIdx.x * 16;

    const int tid  = threadIdx.x;
    const int lane = tid & 31;
    const int wave = tid >> 5;
    const int lrow = lane & 15;
    const int hi8  = (lane & 16) ? 8 : 0;
    const int hi16 = (lane & 16) ? 16 : 0;

    // ---- Phase A: scores = q k^T / 8, masked, into LDS ----
    const __bf16* qrowp = qbf + ((size_t)bh * SEQ + qbase + lrow) * DHEAD;
    v16bf aq0 = load_a_bf16(qrowp + 0  + hi8);   // K = 0..31
    v16bf aq1 = load_a_bf16(qrowp + 32 + hi8);   // K = 32..63

#pragma unroll
    for (int it = 0; it < 8; ++it) {
        const int kt = wave * 8 + it;            // k-tile index (0..63)
        const int n  = kt * 16 + lrow;           // kv position
        const __bf16* kp = kbf + ((size_t)bh * SEQ + n) * DHEAD;
        v16bf b0 = load_b_bf16(kp + hi8);
        v16bf b1 = load_b_bf16(kp + 32 + hi8);
        v8f c = vzero8();
        c = wmma_bf16(aq0, b0, c);
        c = wmma_bf16(aq1, b1, c);
#pragma unroll
        for (int r = 0; r < 8; ++r) {
            const int M  = r + hi8;
            const int qg = qbase + M;
            float v = c[r] * 0.125f;             // 1/sqrt(64)
            if (mask[((size_t)b * SEQ + qg) * SEQ + n]) v = -1e9f;
            sc[M * SEQ + n] = v;
        }
    }
    __syncthreads();

    // ---- Phase B: row softmax; write attn f32 out + bf16 to LDS ----
    const int r  = tid >> 4;    // row 0..15
    const int c0 = tid & 15;    // 16 threads per row
    float m = -3.4e38f;
    for (int j = c0; j < SEQ; j += 16) m = fmaxf(m, sc[r * SEQ + j]);
    red[tid] = m;
    __syncthreads();
    m = -3.4e38f;
#pragma unroll
    for (int i = 0; i < 16; ++i) m = fmaxf(m, red[r * 16 + i]);
    __syncthreads();
    float s = 0.f;
    for (int j = c0; j < SEQ; j += 16) {
        float e = __expf(sc[r * SEQ + j] - m);
        sc[r * SEQ + j] = e;
        s += e;
    }
    red[tid] = s;
    __syncthreads();
    s = 0.f;
#pragma unroll
    for (int i = 0; i < 16; ++i) s += red[r * 16 + i];
    const float inv = 1.f / s;
    float* arow_out = attn_out + ((size_t)bh * SEQ + qbase + r) * SEQ;
    for (int j = c0; j < SEQ; j += 16) {
        float p = sc[r * SEQ + j] * inv;
        arow_out[j]     = p;
        pb[r * SEQ + j] = (__bf16)p;
    }
    __syncthreads();

    // ---- Phase C: context = P @ V, K split across wave pairs ----
    const int tile  = wave & 3;   // output d-tile (0..3)
    const int khalf = wave >> 2;  // K half (0/1)
    const int dim   = tile * 16 + lrow;
    const __bf16* vrow = vtbf + ((size_t)bh * DHEAD + dim) * SEQ + hi8;
    const __bf16* prow = pb + lrow * SEQ + hi8;
    v8f cc = vzero8();
#pragma unroll 4
    for (int kc = khalf * 16; kc < khalf * 16 + 16; ++kc) {
        const int K0 = kc * 32;
        v16bf fa = load_a_bf16(prow + K0);       // ds_load_b128 x2, no cvt
        v16bf bb = load_b_bf16(vrow + K0);
        cc = wmma_bf16(fa, bb, cc);
    }
#pragma unroll
    for (int rr = 0; rr < 8; ++rr)
        psum[(wave * 16 + rr + hi8) * 16 + lrow] = cc[rr];
    __syncthreads();

    for (int idx = tid; idx < 1024; idx += 256) {
        const int tl = idx >> 8, rw = (idx >> 4) & 15, nn = idx & 15;
        float v = psum[(tl * 16 + rw) * 16 + nn] +
                  psum[((tl + 4) * 16 + rw) * 16 + nn];
        const int row = b * SEQ + qbase + rw;         // global token row
        const int col = h * DHEAD + tl * 16 + nn;     // context column
        ctx[(size_t)row * DMODEL + col] = (__bf16)v;
    }
}

// ============================================================
// Kernel 3: output projection + bias + residual + LayerNorm.
// grid = NT/16, 256 threads, dynamic LDS = 16*768*4 = 48KB.
// ============================================================
__global__ __launch_bounds__(256) void out_ln_kernel(
    const __bf16* __restrict__ ctx,
    const __bf16* __restrict__ wto,
    const float* __restrict__ bo,
    const float* __restrict__ Qres,
    const float* __restrict__ gamma,
    const float* __restrict__ beta,
    float* __restrict__ out)
{
    extern __shared__ char smem[];
    float* xsh = (float*)smem;         // [16][768]
    __shared__ float red1[256];
    __shared__ float red2[256];

    const int tid  = threadIdx.x;
    const int lane = tid & 31;
    const int wave = tid >> 5;
    const int lrow = lane & 15;
    const int hi8  = (lane & 16) ? 8 : 0;
    const int hi16 = (lane & 16) ? 16 : 0;
    const int rowbase = blockIdx.x * 16;

    v8f c[6];
#pragma unroll
    for (int i = 0; i < 6; ++i) c[i] = vzero8();

    const __bf16* arow = ctx + (size_t)(rowbase + lrow) * DMODEL + hi8;
#pragma unroll 2
    for (int kc = 0; kc < 24; ++kc) {
        const int K0 = kc * 32;
        v16bf a = load_a_bf16(arow + K0);
#pragma unroll
        for (int i = 0; i < 6; ++i) {
            const int n = (wave * 6 + i) * 16 + lrow;
            v16bf bfr = load_b_bf16(wto + (size_t)n * DMODEL + K0 + hi8);
            c[i] = wmma_bf16(a, bfr, c[i]);
        }
    }

#pragma unroll
    for (int i = 0; i < 6; ++i) {
        const int col = (wave * 6 + i) * 16 + lrow;
        const float bb = bo[col];
#pragma unroll
        for (int r = 0; r < 8; ++r) {
            const int M = r + hi8;
            float v = c[i][r] + bb + Qres[(size_t)(rowbase + M) * DMODEL + col];
            xsh[M * DMODEL + col] = v;
        }
    }
    __syncthreads();

    // LayerNorm: 16 threads per row
    const int r  = tid >> 4;
    const int c0 = tid & 15;
    float s1 = 0.f, s2 = 0.f;
    for (int j = c0; j < DMODEL; j += 16) {
        float x = xsh[r * DMODEL + j];
        s1 += x; s2 += x * x;
    }
    red1[tid] = s1; red2[tid] = s2;
    __syncthreads();
    float mu = 0.f, m2 = 0.f;
#pragma unroll
    for (int i = 0; i < 16; ++i) { mu += red1[r * 16 + i]; m2 += red2[r * 16 + i]; }
    mu *= (1.f / DMODEL);
    float var = m2 * (1.f / DMODEL) - mu * mu;
    float rs = rsqrtf(var + 1e-5f);
    const int grow = rowbase + r;
    for (int j = c0; j < DMODEL; j += 16) {
        float x = xsh[r * DMODEL + j];
        out[(size_t)grow * DMODEL + j] = (x - mu) * rs * gamma[j] + beta[j];
    }
}

// ============================================================
// launcher
// ============================================================
extern "C" void kernel_launch(void* const* d_in, const int* in_sizes, int n_in,
                              void* d_out, int out_size, void* d_ws, size_t ws_size,
                              hipStream_t stream) {
    const float* Q    = (const float*)d_in[0];
    const float* K    = (const float*)d_in[1];
    const float* V    = (const float*)d_in[2];
    const unsigned char* mask = (const unsigned char*)d_in[3]; // bool array
    const float* W_q  = (const float*)d_in[4];
    const float* b_q  = (const float*)d_in[5];
    const float* W_k  = (const float*)d_in[6];
    const float* b_k  = (const float*)d_in[7];
    const float* W_v  = (const float*)d_in[8];
    const float* b_v  = (const float*)d_in[9];
    const float* W_o  = (const float*)d_in[10];
    const float* b_o  = (const float*)d_in[11];
    const float* gamma = (const float*)d_in[12];
    const float* beta  = (const float*)d_in[13];

    // workspace layout (bf16 elements)
    __bf16* wt   = (__bf16*)d_ws;                 // 4 transposed weights
    __bf16* qkv  = wt + (size_t)4 * WELEM;        // q | k | vT
    __bf16* ctx  = qkv + (size_t)3 * QKVE;        // [NT][768]
    __bf16* wto  = wt + (size_t)3 * WELEM;

    float* out_main = (float*)d_out;                           // [NT][768]
    float* attn_out = out_main + (size_t)NT * DMODEL;          // [B,H,S,S]

    prep_weights_kernel<<<(4 * WELEM) / 256, 256, 0, stream>>>(
        W_q, W_k, W_v, W_o, wt);

    qkv_proj_kernel<<<dim3(NT / 64, DMODEL / 64, 3), 256, 0, stream>>>(
        Q, K, V, wt, b_q, b_k, b_v, qkv);

    attn_kernel<<<dim3(SEQ / 16, BATCH * NHEAD), 256,
                  16 * SEQ * (sizeof(float) + sizeof(__bf16)), stream>>>(
        qkv, qkv + QKVE, qkv + (size_t)2 * QKVE, mask, attn_out, ctx);

    out_ln_kernel<<<NT / 16, 256, 16 * DMODEL * sizeof(float), stream>>>(
        ctx, wto, b_o, Q, gamma, beta, out_main);
}
